// Self_Attn_78151224918706
// MI455X (gfx1250) — matrix operand order repaired and checked by reading, not compiled, not run on hardware
//
#include <hip/hip_runtime.h>
#include <hip/hip_bf16.h>
#include <math.h>

typedef __attribute__((ext_vector_type(16))) __bf16 v16bf;
typedef __attribute__((ext_vector_type(8)))  __bf16 v8bf;
typedef __attribute__((ext_vector_type(8)))  float  v8f;
typedef __attribute__((ext_vector_type(4)))  unsigned int u32x4;
typedef __attribute__((ext_vector_type(8)))  unsigned int u32x8;

#define WMMA_BF16(A, B, C) \
    __builtin_amdgcn_wmma_f32_16x16x32_bf16(false, (A), false, (B), (short)0, (C), false, false)

#define HW   65536   // 256*256
#define DIM  128
#define KV   1089    // 33*33
#define NB   2
#define KBYTES (KV * DIM * 2)   // 278784 bytes of bf16 K per batch (fits 320KB LDS)

static __device__ __forceinline__ int imin(int a, int b) { return a < b ? a : b; }

// ---------------- bilinear helpers (align_corners=True) ----------------
static __device__ __forceinline__ float bilin_f32(const float* __restrict__ src, int H, int W,
                                                  float fy, float fx) {
    int y0 = (int)fy; y0 = imin(y0, H - 1); int y1 = imin(y0 + 1, H - 1);
    int x0 = (int)fx; x0 = imin(x0, W - 1); int x1 = imin(x0 + 1, W - 1);
    float wy = fy - (float)y0, wx = fx - (float)x0;
    float a = src[y0 * W + x0], b = src[y0 * W + x1];
    float c = src[y1 * W + x0], d = src[y1 * W + x1];
    float top = a + (b - a) * wx;
    float bot = c + (d - c) * wx;
    return top + (bot - top) * wy;
}

static __device__ __forceinline__ float bilin_bf16(const __bf16* __restrict__ src, int H, int W,
                                                   float fy, float fx) {
    int y0 = (int)fy; y0 = imin(y0, H - 1); int y1 = imin(y0 + 1, H - 1);
    int x0 = (int)fx; x0 = imin(x0, W - 1); int x1 = imin(x0 + 1, W - 1);
    float wy = fy - (float)y0, wx = fx - (float)x0;
    float a = (float)src[y0 * W + x0], b = (float)src[y0 * W + x1];
    float c = (float)src[y1 * W + x0], d = (float)src[y1 * W + x1];
    float top = a + (b - a) * wx;
    float bot = c + (d - c) * wx;
    return top + (bot - top) * wy;
}

// ---------------- WMMA fragment loaders ----------------
// A (16x32 bf16): lane<16 -> row=lane,   K = {kb+0..7, kb+16..23}
//                 lane>=16 -> row=lane-16, K = {kb+8..15, kb+24..31}
static __device__ __forceinline__ v16bf load_a_bf16(const __bf16* __restrict__ rowptr,
                                                    int kb, int half) {
    const v8bf lo = *reinterpret_cast<const v8bf*>(rowptr + kb + half * 8);
    const v8bf hi = *reinterpret_cast<const v8bf*>(rowptr + kb + 16 + half * 8);
    v16bf r;
#pragma unroll
    for (int i = 0; i < 8; ++i) { r[i] = lo[i]; r[i + 8] = hi[i]; }
    return r;
}

static __device__ __forceinline__ v16bf load_a_f32(const float* __restrict__ rowptr,
                                                   int kb, int half) {
    const float* p0 = rowptr + kb + half * 8;
    const float* p1 = rowptr + kb + 16 + half * 8;
    v16bf r;
#pragma unroll
    for (int i = 0; i < 8; ++i) { r[i] = (__bf16)p0[i]; r[i + 8] = (__bf16)p1[i]; }
    return r;
}

// B (32x16 bf16): lane<16 -> col=lane,   K = {kb+0..7, kb+16..23} ; lane>=16 -> col=lane-16, K = {kb+8..15, kb+24..31}
// "colrowptr" points at the contiguous 128-elem source row for this lane's column.
static __device__ __forceinline__ v16bf load_b_bf16(const __bf16* colrowptr, int kb, int half) {
    const v8bf lo = *reinterpret_cast<const v8bf*>(colrowptr + kb + half * 8);
    const v8bf hi = *reinterpret_cast<const v8bf*>(colrowptr + kb + 16 + half * 8);
    v16bf r;
#pragma unroll
    for (int i = 0; i < 8; ++i) { r[i] = lo[i]; r[i + 8] = hi[i]; }
    return r;
}

// ---------------- K1: upsample conv_a/conv_b -> merged (pixel-major bf16) ----------------
__global__ __launch_bounds__(128) void upsample_merge_kernel(
    const float* __restrict__ ca, const float* __restrict__ cb, __bf16* __restrict__ merged) {
    const int p = blockIdx.x;            // pixel 0..65535
    const int b = blockIdx.y;            // batch
    const int c = threadIdx.x;           // channel 0..127
    const int y = p >> 8, x = p & 255;
    float v;
    if (c < 64) {
        const float s = 63.0f / 255.0f;
        v = bilin_f32(ca + ((size_t)b * 64 + c) * 4096, 64, 64, y * s, x * s);
    } else {
        const float s = 127.0f / 255.0f;
        v = bilin_f32(cb + ((size_t)b * 64 + (c - 64)) * 16384, 128, 128, y * s, x * s);
    }
    merged[((size_t)b * HW + p) * DIM + c] = (__bf16)v;
}

// ---------------- K2: qk = W1 x merged^T  (WMMA, out channel-major bf16 == Q matrix) ------
__global__ __launch_bounds__(128) void qk_gemm_kernel(
    const float* __restrict__ W1, const __bf16* __restrict__ merged, __bf16* __restrict__ qk) {
    const int lane = threadIdx.x & 31;
    const int wid  = threadIdx.x >> 5;
    const int half = lane >> 4;
    const int lr   = lane & 15;
    const int t     = blockIdx.x * 4 + wid;        // 0..65535 wave tiles
    const int ptile = t & 4095;
    const int otile = (t >> 12) & 7;
    const int b     = t >> 15;
    const int p0 = ptile << 4, o0 = otile << 4;

    const float*  wrow = W1 + (size_t)(o0 + lr) * DIM;
    const __bf16* mrow = merged + ((size_t)b * HW + (p0 + lr)) * DIM;

    v8f acc = {0.f, 0.f, 0.f, 0.f, 0.f, 0.f, 0.f, 0.f};
#pragma unroll
    for (int f = 0; f < 4; ++f) {
        v16bf a  = load_a_f32(wrow, f * 32, half);
        v16bf bb = load_b_bf16(mrow, f * 32, half);
        acc = WMMA_BF16(a, bb, acc);
    }
    __bf16* outb = qk + (size_t)b * HW * DIM;
#pragma unroll
    for (int r = 0; r < 8; ++r)
        outb[(size_t)(o0 + r + half * 8) * HW + p0 + lr] = (__bf16)acc[r];
}

// ---------------- K3: resize qk (per channel, 256x256 -> 33x33), flat == K rows ----------
__global__ __launch_bounds__(256) void resize_k_kernel(
    const __bf16* __restrict__ qk, __bf16* __restrict__ kbf) {
    const int i = blockIdx.x * 256 + threadIdx.x;         // 0 .. 2*139392-1 (exact)
    const int b = i / (KV * DIM);
    const int j = i - b * (KV * DIM);
    const int o = j / KV;
    const int r = j - o * KV;
    const int yk = r / 33, xk = r - (r / 33) * 33;
    const float s = 255.0f / 32.0f;
    float v = bilin_bf16(qk + ((size_t)b * DIM + o) * HW, 256, 256, yk * s, xk * s);
    kbf[i] = (__bf16)v;
}

// ---------------- K4: resize logits -> value, flat == [1089 x 2] rows --------------------
__global__ __launch_bounds__(256) void resize_v_kernel(
    const float* __restrict__ logits, float* __restrict__ val) {
    const int i = blockIdx.x * 256 + threadIdx.x;         // 0 .. 2*2178-1
    if (i >= NB * 2 * KV) return;
    const int b = i / (2 * KV);
    const int j = i - b * (2 * KV);
    const int o = j / KV;
    const int r = j - o * KV;
    const int yk = r / 33, xk = r - (r / 33) * 33;
    const float s = 255.0f / 32.0f;
    val[i] = bilin_f32(logits + ((size_t)b * 2 + o) * HW, 256, 256, yk * s, xk * s);
}

// ---------------- K5: flash attention with TDM-staged K in LDS --------------------------
// Block = 512 threads (16 waves) sharing one batch's K (278784 B) in LDS via the
// Tensor Data Mover; each wave owns a 16-row Q tile held in A-fragments.
__global__ __launch_bounds__(512) void attn_kernel(
    const __bf16* __restrict__ qbf, const __bf16* __restrict__ kbf,
    const float* __restrict__ val, float* __restrict__ att) {
    __shared__ __align__(16) char smem[KBYTES];   // single LDS object -> offset 0

    const int lane = threadIdx.x & 31;
    const int wid  = threadIdx.x >> 5;            // 0..15
    const int half = lane >> 4;
    const int lr   = lane & 15;
    const int b    = blockIdx.x >> 8;             // 512 blocks: 256 per batch
    const int q0   = (((blockIdx.x & 255) << 4) + wid) << 4;   // 16 q rows per wave

    // --- TDM: DMA the whole K matrix [1089 rows x 256B] into LDS (one 2D tile) ---
    if (wid == 0) {
        const unsigned long long ga =
            (unsigned long long)(const void*)(kbf + (size_t)b * KV * DIM);
        u32x4 g0;
        g0[0] = 1u;                                            // count=1, user desc
        g0[1] = 0u;                                            // lds_addr = 0
        g0[2] = (unsigned int)ga;                              // global_addr[31:0]
        g0[3] = (unsigned int)((ga >> 32) & 0x01FFFFFFu)       // global_addr[56:32]
              | 0x80000000u;                                   // type = 2 (image)
        u32x8 g1;
        g1[0] = 3u << 16;            // workgroup_mask=0, data_size=3 (8 bytes)
        g1[1] = 32u << 16;           // atomic_barrier_addr=0 | tensor_dim0.lo16 = 32
        g1[2] = 1089u << 16;         // tensor_dim0.hi16 = 0 | tensor_dim1.lo16 = 1089
        g1[3] = 32u << 16;           // tensor_dim1.hi16 = 0 | tile_dim0 = 32
        g1[4] = 1089u;               // tile_dim1 = 1089 | tile_dim2 = 0
        g1[5] = 32u;                 // tensor_dim0_stride.lo32 = 32 (elements of 8B)
        g1[6] = 0u;                  // stride0.hi16 | stride1.lo16
        g1[7] = 0u;                  // stride1.hi32
        asm volatile("tensor_load_to_lds %0, %1" :: "s"(g0), "s"(g1) : "memory");
        __builtin_amdgcn_s_wait_tensorcnt(0);
    }
    __syncthreads();
    const __bf16* ksh = (const __bf16*)smem;

    const __bf16* qrow = qbf + (size_t)b * HW * DIM + (size_t)(q0 + lr) * DIM;
    v16bf a0 = load_a_bf16(qrow, 0,  half);
    v16bf a1 = load_a_bf16(qrow, 32, half);
    v16bf a2 = load_a_bf16(qrow, 64, half);
    v16bf a3 = load_a_bf16(qrow, 96, half);

    float rm[8], rl[8], ro0[8], ro1[8];
#pragma unroll
    for (int r = 0; r < 8; ++r) { rm[r] = -1e30f; rl[r] = 0.f; ro0[r] = 0.f; ro1[r] = 0.f; }

    const float* vb = val + (size_t)b * KV * 2;
    const float scale = 0.08838834764831845f;   // 1/sqrt(128)

    for (int t = 0; t < 69; ++t) {
        const int  n     = t * 16 + lr;
        const bool valid = (n < KV);
        const int  nc    = valid ? n : (KV - 1);
        const __bf16* krow = ksh + (size_t)nc * DIM;

        v8f acc = {0.f, 0.f, 0.f, 0.f, 0.f, 0.f, 0.f, 0.f};
        acc = WMMA_BF16(a0, load_b_bf16(krow, 0,  half), acc);
        acc = WMMA_BF16(a1, load_b_bf16(krow, 32, half), acc);
        acc = WMMA_BF16(a2, load_b_bf16(krow, 64, half), acc);
        acc = WMMA_BF16(a3, load_b_bf16(krow, 96, half), acc);

        const float v0 = vb[nc * 2 + 0];
        const float v1 = vb[nc * 2 + 1];

#pragma unroll
        for (int r = 0; r < 8; ++r) {
            float s = valid ? acc[r] * scale : -1e30f;
            float tm = s;
            tm = fmaxf(tm, __shfl_xor(tm, 1, 16));
            tm = fmaxf(tm, __shfl_xor(tm, 2, 16));
            tm = fmaxf(tm, __shfl_xor(tm, 4, 16));
            tm = fmaxf(tm, __shfl_xor(tm, 8, 16));
            const float nm    = fmaxf(rm[r], tm);
            const float alpha = __expf(rm[r] - nm);
            const float p     = valid ? __expf(s - nm) : 0.f;
            float ps = p, p0 = p * v0, p1 = p * v1;
            ps += __shfl_xor(ps, 1, 16); p0 += __shfl_xor(p0, 1, 16); p1 += __shfl_xor(p1, 1, 16);
            ps += __shfl_xor(ps, 2, 16); p0 += __shfl_xor(p0, 2, 16); p1 += __shfl_xor(p1, 2, 16);
            ps += __shfl_xor(ps, 4, 16); p0 += __shfl_xor(p0, 4, 16); p1 += __shfl_xor(p1, 4, 16);
            ps += __shfl_xor(ps, 8, 16); p0 += __shfl_xor(p0, 8, 16); p1 += __shfl_xor(p1, 8, 16);
            rl[r]  = rl[r]  * alpha + ps;
            ro0[r] = ro0[r] * alpha + p0;
            ro1[r] = ro1[r] * alpha + p1;
            rm[r]  = nm;
        }
    }

    if (lr < 2) {
#pragma unroll
        for (int r = 0; r < 8; ++r) {
            const int qn = q0 + r + half * 8;
            const float o = (lr == 0 ? ro0[r] : ro1[r]) / rl[r];
            att[(size_t)b * HW * 2 + (size_t)qn * 2 + lr] = o;  // [B,65536,2]
        }
    }
}

// ---------------- K6: residual + W2 (1x1) + per-block partial BN stats ------------------
__global__ __launch_bounds__(256) void conv2_stats_kernel(
    const float* __restrict__ att, const float* __restrict__ logits,
    const float* __restrict__ W2, float* __restrict__ outpre, float* __restrict__ partials) {
    const int idx = blockIdx.x * 256 + threadIdx.x;  // 0..131071 = (b,pix)
    const int b   = idx >> 16;
    const int pix = idx & 65535;
    const size_t base = (size_t)b * 131072;
    // raw reshape of [B,65536,2] back to [B,2,256,256] is identity on the flat buffer:
    const float A0 = att[base + pix]         + logits[base + pix];
    const float A1 = att[base + 65536 + pix] + logits[base + 65536 + pix];
    const float o0 = W2[0] * A0 + W2[1] * A1;
    const float o1 = W2[2] * A0 + W2[3] * A1;
    outpre[base + pix]         = o0;
    outpre[base + 65536 + pix] = o1;

    float s0 = o0, s1 = o1, q0 = o0 * o0, q1 = o1 * o1;
#pragma unroll
    for (int m = 1; m < 32; m <<= 1) {
        s0 += __shfl_xor(s0, m, 32); s1 += __shfl_xor(s1, m, 32);
        q0 += __shfl_xor(q0, m, 32); q1 += __shfl_xor(q1, m, 32);
    }
    __shared__ float sh[8 * 4];
    const int wid = threadIdx.x >> 5;
    if ((threadIdx.x & 31) == 0) {
        sh[wid * 4 + 0] = s0; sh[wid * 4 + 1] = s1;
        sh[wid * 4 + 2] = q0; sh[wid * 4 + 3] = q1;
    }
    __syncthreads();
    if (threadIdx.x < 4) {
        float t = 0.f;
#pragma unroll
        for (int w = 0; w < 8; ++w) t += sh[w * 4 + threadIdx.x];
        partials[blockIdx.x * 4 + threadIdx.x] = t;
    }
}

// ---------------- K7: deterministic fixed-order reduction of 512 partials ---------------
__global__ void reduce_stats_kernel(const float* __restrict__ partials, float* __restrict__ stats) {
    const int k = threadIdx.x;
    if (k < 4) {
        float s = 0.f;
        for (int i = 0; i < 512; ++i) s += partials[i * 4 + k];
        stats[k] = s;
    }
}

// ---------------- K8: BN normalize -----------------------------------------------------
__global__ __launch_bounds__(256) void bn_finalize_kernel(
    const float* __restrict__ outpre, const float* __restrict__ stats,
    const float* __restrict__ gamma, const float* __restrict__ beta,
    float* __restrict__ out) {
    const int idx = blockIdx.x * 256 + threadIdx.x;   // 0..262143
    const int o = (idx >> 16) & 1;                    // channel
    const float inv_n = 1.0f / 131072.0f;
    const float mean = stats[o] * inv_n;
    const float var  = stats[2 + o] * inv_n - mean * mean;
    out[idx] = (outpre[idx] - mean) * rsqrtf(var + 1e-5f) * gamma[o] + beta[o];
}

// =======================================================================================
extern "C" void kernel_launch(void* const* d_in, const int* in_sizes, int n_in,
                              void* d_out, int out_size, void* d_ws, size_t ws_size,
                              hipStream_t stream) {
    (void)in_sizes; (void)n_in; (void)out_size; (void)ws_size;
    const float* conv_a = (const float*)d_in[0];
    const float* conv_b = (const float*)d_in[1];
    const float* logits = (const float*)d_in[2];
    const float* W1     = (const float*)d_in[3];
    const float* W2     = (const float*)d_in[4];
    const float* gamma  = (const float*)d_in[5];
    const float* beta   = (const float*)d_in[6];
    float* out = (float*)d_out;

    char* ws = (char*)d_ws;
    size_t off = 0;
    auto bump = [&](size_t bytes) -> char* {
        char* p = ws + off;
        off += (bytes + 255) & ~(size_t)255;
        return p;
    };
    __bf16* merged   = (__bf16*)bump((size_t)NB * HW * DIM * 2);   // 33.5 MB
    __bf16* qkbf     = (__bf16*)bump((size_t)NB * HW * DIM * 2);   // 33.5 MB
    __bf16* kbf      = (__bf16*)bump((size_t)NB * KV * DIM * 2);   // 557 KB
    float*  val      = (float*) bump((size_t)NB * KV * 2 * 4);
    float*  att      = (float*) bump((size_t)NB * HW * 2 * 4);     // 1 MB
    float*  outpre   = (float*) bump((size_t)NB * HW * 2 * 4);     // 1 MB
    float*  partials = (float*) bump(512 * 4 * 4);
    float*  stats    = (float*) bump(4 * 4);

    upsample_merge_kernel<<<dim3(HW, NB), 128, 0, stream>>>(conv_a, conv_b, merged);
    qk_gemm_kernel<<<16384, 128, 0, stream>>>(W1, merged, qkbf);
    resize_k_kernel<<<(NB * KV * DIM) / 256, 256, 0, stream>>>(qkbf, kbf);
    resize_v_kernel<<<(NB * 2 * KV + 255) / 256, 256, 0, stream>>>(logits, val);
    attn_kernel<<<512, 512, 0, stream>>>(qkbf, kbf, val, att);
    conv2_stats_kernel<<<512, 256, 0, stream>>>(att, logits, W2, outpre, partials);
    reduce_stats_kernel<<<1, 32, 0, stream>>>(partials, stats);
    bn_finalize_kernel<<<1024, 256, 0, stream>>>(outpre, stats, gamma, beta, out);
}
